// JetBlock_74620761801297
// MI455X (gfx1250) — compile-verified, matrix operand order checked by reading, not run on hardware
//
#include <hip/hip_runtime.h>
#include <hip/hip_bf16.h>

typedef __attribute__((ext_vector_type(8)))  __bf16 v8bf;
typedef __attribute__((ext_vector_type(16))) __bf16 v16bf;
typedef __attribute__((ext_vector_type(8)))  float  v8f;

constexpr int SEQ = 2048;
constexpr int HID = 2048;
constexpr int NH  = 8;
constexpr int DK  = 96;
constexpr int DV  = 192;
constexpr int TK  = NH * DK;      // 768
constexpr int TV  = NH * DV;      // 1536

// --------------------------------------------------------------------------
// CDNA5 async global->LDS copy (ASYNCcnt), guarded so either toolchain builds.
// Builtin signature: (global int4* src, lds int4* dst, imm offset, imm cpol).
// --------------------------------------------------------------------------
#if defined(__HIP_DEVICE_COMPILE__) && \
    __has_builtin(__builtin_amdgcn_global_load_async_to_lds_b128) && \
    __has_builtin(__builtin_amdgcn_s_wait_asynccnt)
#define USE_ASYNC_LDS 1
#else
#define USE_ASYNC_LDS 0
#endif

typedef int vsi4 __attribute__((__vector_size__(16)));
#if USE_ASYNC_LDS
typedef __attribute__((address_space(1))) vsi4 g_vsi4;   // global int4
typedef __attribute__((address_space(3))) vsi4 l_vsi4;   // LDS int4
#endif

__device__ __forceinline__ void cp16_g2l(__bf16* lds_dst, const __bf16* gsrc) {
#if USE_ASYNC_LDS
    __builtin_amdgcn_global_load_async_to_lds_b128(
        (g_vsi4*)(uintptr_t)gsrc,
        (l_vsi4*)(unsigned)(uintptr_t)lds_dst,
        0, 0);
#else
    *(v8bf*)lds_dst = *(const v8bf*)gsrc;
#endif
}

// Wait until at most the newest tile's 10 copies are outstanding.
__device__ __forceinline__ void wait_async_pending() {
#if USE_ASYNC_LDS
    __builtin_amdgcn_s_wait_asynccnt(10);
#endif
}
__device__ __forceinline__ void wait_async_all() {
#if USE_ASYNC_LDS
    __builtin_amdgcn_s_wait_asynccnt(0);
#endif
}

__device__ __forceinline__ float fast_silu(float x) {
    return x / (1.0f + __expf(-x));
}

__device__ __forceinline__ float wave_sum(float x) {
#pragma unroll
    for (int off = 16; off > 0; off >>= 1)
        x += __shfl_xor(x, off, 32);
    return x;
}

// Build a 16x32 bf16 WMMA fragment from an LDS row.
// lane<16 -> K kb+0..7 / kb+16..23 ; lane>=16 -> K kb+8..15 / kb+24..31.
__device__ __forceinline__ v16bf lds_frag(const __bf16* row, int kb) {
    v8bf lo = *(const v8bf*)(row + kb);
    v8bf hi = *(const v8bf*)(row + kb + 16);
    return __builtin_shufflevector(lo, hi, 0, 1, 2, 3, 4, 5, 6, 7,
                                           8, 9, 10, 11, 12, 13, 14, 15);
}

__device__ __forceinline__ void store_tile(float* __restrict__ C, int N,
                                           int row, int col, v8f acc, int act) {
#pragma unroll
    for (int i = 0; i < 8; ++i) {
        float x = acc[i];
        if (act) x = fast_silu(x);
        C[(size_t)(row + i) * N + col] = x;
    }
}

// -------------------------------------------------------------------------
// f32 -> bf16 conversion (grid-stride)
// -------------------------------------------------------------------------
__global__ void cvt_kernel(const float* __restrict__ in, __bf16* __restrict__ out, int n) {
    int i = blockIdx.x * blockDim.x + threadIdx.x;
    int stride = gridDim.x * blockDim.x;
    for (; i < n; i += stride) out[i] = (__bf16)in[i];
}

// -------------------------------------------------------------------------
// C[M,N] = act(A[M,K] * B[N,K]^T), bf16 inputs, f32 output.
// 256 threads (8 waves as 2x4). Block tile 64x256, wave tile 32x64, K-step 64.
// Double-buffered LDS staging via async global->LDS copies (ASYNCcnt).
// -------------------------------------------------------------------------
__global__ __launch_bounds__(256) void gemm_bf16_nt(
    const __bf16* __restrict__ A, const __bf16* __restrict__ B,
    float* __restrict__ C, int M, int N, int K, int act) {
    __shared__ __bf16 lA[2][64][72];     // 64 K cols, padded to 72
    __shared__ __bf16 lB[2][256][72];

    const int tid   = threadIdx.x;
    const int lane  = tid & 31;
    const int wave  = tid >> 5;
    const int wm    = wave & 1;          // M offset wm*32
    const int wn    = wave >> 1;         // N offset wn*64
    const int mBase = blockIdx.x * 64;
    const int nBase = blockIdx.y * 256;

    const int rsel  = lane & 15;
    const int khalf = (lane >> 4) << 3;  // 0 or 8

    v8f acc[2][4];
#pragma unroll
    for (int mt = 0; mt < 2; ++mt)
#pragma unroll
        for (int nt = 0; nt < 4; ++nt) acc[mt][nt] = (v8f){};

    // Staging: 16B chunks; chunk c -> row c>>3, col (c&7)*8.
    auto prefetch = [&](int buf, int k0) {
#pragma unroll
        for (int j = 0; j < 2; ++j) {            // A: 64x64 -> 512 chunks
            int c = tid + 256 * j;
            int r = c >> 3, cs = (c & 7) << 3;
            cp16_g2l(&lA[buf][r][cs], A + (size_t)(mBase + r) * K + k0 + cs);
        }
#pragma unroll
        for (int j = 0; j < 8; ++j) {            // B: 256x64 -> 2048 chunks
            int c = tid + 256 * j;
            int r = c >> 3, cs = (c & 7) << 3;
            cp16_g2l(&lB[buf][r][cs], B + (size_t)(nBase + r) * K + k0 + cs);
        }
    };

    const int niter = K >> 6;    // K / 64
    prefetch(0, 0);

    for (int it = 0; it < niter; ++it) {
        const int cur = it & 1;
        if (it + 1 < niter) {
            prefetch(cur ^ 1, (it + 1) << 6);
            wait_async_pending();   // current tile's copies have completed
        } else {
            wait_async_all();
        }
        __syncthreads();

#pragma unroll
        for (int kk = 0; kk < 2; ++kk) {         // two 16x16x32 k-slices
            const int kb = (kk << 5) + khalf;
            v16bf a0 = lds_frag(&lA[cur][wm * 32 + rsel][0],      kb);
            v16bf a1 = lds_frag(&lA[cur][wm * 32 + 16 + rsel][0], kb);
            v16bf b0 = lds_frag(&lB[cur][wn * 64 + rsel][0],      kb);
            v16bf b1 = lds_frag(&lB[cur][wn * 64 + 16 + rsel][0], kb);
            v16bf b2 = lds_frag(&lB[cur][wn * 64 + 32 + rsel][0], kb);
            v16bf b3 = lds_frag(&lB[cur][wn * 64 + 48 + rsel][0], kb);

            acc[0][0] = __builtin_amdgcn_wmma_f32_16x16x32_bf16(false, a0, false, b0,
                            (short)0, acc[0][0], false, false);
            acc[0][1] = __builtin_amdgcn_wmma_f32_16x16x32_bf16(false, a0, false, b1,
                            (short)0, acc[0][1], false, false);
            acc[0][2] = __builtin_amdgcn_wmma_f32_16x16x32_bf16(false, a0, false, b2,
                            (short)0, acc[0][2], false, false);
            acc[0][3] = __builtin_amdgcn_wmma_f32_16x16x32_bf16(false, a0, false, b3,
                            (short)0, acc[0][3], false, false);
            acc[1][0] = __builtin_amdgcn_wmma_f32_16x16x32_bf16(false, a1, false, b0,
                            (short)0, acc[1][0], false, false);
            acc[1][1] = __builtin_amdgcn_wmma_f32_16x16x32_bf16(false, a1, false, b1,
                            (short)0, acc[1][1], false, false);
            acc[1][2] = __builtin_amdgcn_wmma_f32_16x16x32_bf16(false, a1, false, b2,
                            (short)0, acc[1][2], false, false);
            acc[1][3] = __builtin_amdgcn_wmma_f32_16x16x32_bf16(false, a1, false, b3,
                            (short)0, acc[1][3], false, false);
        }
        __syncthreads();
    }

    // 16x16 f32 C/D layout: lane<16 -> N=lane, M=vgpr 0..7; lane>=16 -> M=8+vgpr
    const int colB = nBase + wn * 64 + (lane & 15);
    const int rowB = mBase + wm * 32 + ((lane >> 4) << 3);
#pragma unroll
    for (int mt = 0; mt < 2; ++mt)
#pragma unroll
        for (int nt = 0; nt < 4; ++nt)
            store_tile(C, N, rowB + mt * 16, colB + nt * 16, acc[mt][nt], act);
}

// -------------------------------------------------------------------------
// alpha/beta: per (s,h) dot products over HID in f32, one wave per (s,h).
// -------------------------------------------------------------------------
__global__ __launch_bounds__(256) void ab_kernel(
    const float* __restrict__ X, const float* __restrict__ Wa,
    const float* __restrict__ Wb, const float* __restrict__ A_log,
    const float* __restrict__ dt_bias, float* __restrict__ alpha,
    float* __restrict__ beta) {
    const int s    = blockIdx.x;
    const int h    = threadIdx.x >> 5;
    const int lane = threadIdx.x & 31;
    const float* x  = X  + (size_t)s * HID;
    const float* wa = Wa + (size_t)h * HID;
    const float* wb = Wb + (size_t)h * HID;
    float sa = 0.f, sb = 0.f;
    for (int i = lane; i < HID; i += 32) {
        float xv = x[i];
        sa += xv * wa[i];
        sb += xv * wb[i];
    }
    sa = wave_sum(sa);
    sb = wave_sum(sb);
    if (lane == 0) {
        float z  = sa + dt_bias[h];
        float sp = (z > 20.f) ? z : log1pf(__expf(z));
        alpha[s * NH + h] = __expf(-__expf(A_log[h]) * sp);
        beta[s * NH + h]  = 1.f / (1.f + __expf(-sb));
    }
}

// -------------------------------------------------------------------------
// Gated delta scan. Each v-column recurrence is independent given shared
// k_t, q_t, alpha_t, beta_t. One wave per v-column (state = 96 f32 = 3/lane).
// 8 columns of one head per block share LDS-staged q/k/alpha/beta chunks.
// -------------------------------------------------------------------------
__global__ __launch_bounds__(256) void scan_kernel(
    const float* __restrict__ qk,    // [S, 2*TK]  (q | k), post-silu
    const float* __restrict__ vg,    // [S, 2*TV]  (v | g)
    const float* __restrict__ alpha, const float* __restrict__ beta,
    float* __restrict__ o) {         // [S, TV]
    __shared__ float sq[32][DK];
    __shared__ float sk[32][DK];
    __shared__ float sa[32], sb[32];

    const int tid  = threadIdx.x;
    const int lane = tid & 31;
    const int wave = tid >> 5;
    const int head = blockIdx.x / (DV / 8);
    const int col  = (blockIdx.x % (DV / 8)) * 8 + wave;

    const int qoff = head * DK;
    const int koff = TK + head * DK;

    float s0 = 0.f, s1 = 0.f, s2 = 0.f;

    for (int c0 = 0; c0 < SEQ; c0 += 32) {
        __syncthreads();
        for (int i = tid; i < 32 * DK; i += 256) {
            int t = i / DK, kk = i % DK;
            const float* row = qk + (size_t)(c0 + t) * (2 * TK);
            sq[t][kk] = row[qoff + kk];
            sk[t][kk] = row[koff + kk];
        }
        if (tid < 32)      sa[tid]      = alpha[(c0 + tid) * NH + head];
        else if (tid < 64) sb[tid - 32] = beta[(c0 + tid - 32) * NH + head];
        __syncthreads();

        for (int t = 0; t < 32; ++t) {
            float k0v = sk[t][lane];
            float k1v = sk[t][lane + 32];
            float k2v = sk[t][lane + 64];
            float pr = k0v * s0 + k1v * s1 + k2v * s2;
            pr = wave_sum(pr);
            float al = sa[t], be = sb[t];
            float vt = vg[(size_t)(c0 + t) * (2 * TV) + head * DV + col];
            float err = be * (vt - al * pr);      // pred uses decayed state
            s0 = al * s0 + k0v * err;
            s1 = al * s1 + k1v * err;
            s2 = al * s2 + k2v * err;
            float ot = sq[t][lane] * s0 + sq[t][lane + 32] * s1 + sq[t][lane + 64] * s2;
            ot = wave_sum(ot);
            if (lane == 0)
                o[(size_t)(c0 + t) * TV + head * DV + col] = ot;
        }
    }
}

// -------------------------------------------------------------------------
// Gated RMSNorm over DV + silu(g) gate, emit bf16 for the final GEMM.
// One wave per (s,h); lane owns 6 of 192 elements.
// -------------------------------------------------------------------------
__global__ __launch_bounds__(256) void norm_kernel(
    const float* __restrict__ o, const float* __restrict__ vg,
    const float* __restrict__ nw, __bf16* __restrict__ obf) {
    const int lane = threadIdx.x & 31;
    const int wave = threadIdx.x >> 5;
    const int idx  = blockIdx.x * 8 + wave;  // s*NH + h
    const int s = idx >> 3;
    const int h = idx & 7;
    const float* orow = o  + (size_t)s * TV + h * DV;
    const float* grow = vg + (size_t)s * (2 * TV) + TV + h * DV;
    float x[6];
    float ss = 0.f;
#pragma unroll
    for (int j = 0; j < 6; ++j) {
        x[j] = orow[lane + 32 * j];
        ss += x[j] * x[j];
    }
    ss = wave_sum(ss);
    float rms = rsqrtf(ss / (float)DV + 1e-5f);
#pragma unroll
    for (int j = 0; j < 6; ++j) {
        int v = lane + 32 * j;
        float y = x[j] * rms * nw[v] * fast_silu(grow[v]);
        obf[(size_t)s * TV + h * DV + v] = (__bf16)y;
    }
}

// -------------------------------------------------------------------------
extern "C" void kernel_launch(void* const* d_in, const int* in_sizes, int n_in,
                              void* d_out, int out_size, void* d_ws, size_t ws_size,
                              hipStream_t stream) {
    const float* X       = (const float*)d_in[0];
    const float* Wq      = (const float*)d_in[1];
    const float* Wk      = (const float*)d_in[2];
    const float* Wv      = (const float*)d_in[3];
    const float* Wa      = (const float*)d_in[4];
    const float* Wb      = (const float*)d_in[5];
    const float* Wg      = (const float*)d_in[6];
    const float* Wo      = (const float*)d_in[7];
    const float* A_log   = (const float*)d_in[8];
    const float* dt_bias = (const float*)d_in[9];
    const float* norm_w  = (const float*)d_in[10];

    char* ws = (char*)d_ws;
    size_t off = 0;
    auto take = [&](size_t bytes) -> char* {
        char* p = ws + off;
        off += (bytes + 255) & ~(size_t)255;
        return p;
    };

    __bf16* Xbf   = (__bf16*)take((size_t)SEQ * HID * 2);
    __bf16* WqkBf = (__bf16*)take((size_t)2 * TK * HID * 2);   // Wq | Wk packed
    __bf16* WvgBf = (__bf16*)take((size_t)2 * TV * HID * 2);   // Wv | Wg packed
    __bf16* WoBf  = (__bf16*)take((size_t)HID * TV * 2);
    float*  qk    = (float*)take((size_t)SEQ * 2 * TK * 4);    // silu(q)|silu(k)
    float*  vg    = (float*)take((size_t)SEQ * 2 * TV * 4);    // v|g
    float*  alpha = (float*)take((size_t)SEQ * NH * 4);
    float*  beta  = (float*)take((size_t)SEQ * NH * 4);
    float*  o     = (float*)take((size_t)SEQ * TV * 4);
    __bf16* obf   = (__bf16*)take((size_t)SEQ * TV * 2);

    // 1) f32 -> bf16 conversions (X + packed weights)
    cvt_kernel<<<1024, 256, 0, stream>>>(X, Xbf, SEQ * HID);
    cvt_kernel<<<1024, 256, 0, stream>>>(Wq, WqkBf, TK * HID);
    cvt_kernel<<<1024, 256, 0, stream>>>(Wk, WqkBf + (size_t)TK * HID, TK * HID);
    cvt_kernel<<<1024, 256, 0, stream>>>(Wv, WvgBf, TV * HID);
    cvt_kernel<<<1024, 256, 0, stream>>>(Wg, WvgBf + (size_t)TV * HID, TV * HID);
    cvt_kernel<<<1024, 256, 0, stream>>>(Wo, WoBf, HID * TV);

    // 2) alpha/beta (small dots, full f32)
    ab_kernel<<<SEQ, 256, 0, stream>>>(X, Wa, Wb, A_log, dt_bias, alpha, beta);

    // 3) fused q|k projection with silu  (M=2048, N=1536, K=2048)
    gemm_bf16_nt<<<dim3(SEQ / 64, (2 * TK) / 256), 256, 0, stream>>>(
        Xbf, WqkBf, qk, SEQ, 2 * TK, HID, 1);

    // 4) fused v|g projection           (M=2048, N=3072, K=2048)
    gemm_bf16_nt<<<dim3(SEQ / 64, (2 * TV) / 256), 256, 0, stream>>>(
        Xbf, WvgBf, vg, SEQ, 2 * TV, HID, 0);

    // 5) gated delta scan: one wave per (head, v-column)
    scan_kernel<<<NH * (DV / 8), 256, 0, stream>>>(qk, vg, alpha, beta, o);

    // 6) gated RMSNorm -> bf16
    norm_kernel<<<(SEQ * NH) / 8, 256, 0, stream>>>(o, vg, norm_w, obf);

    // 7) output projection               (M=2048, N=2048, K=1536)
    gemm_bf16_nt<<<dim3(SEQ / 64, HID / 256), 256, 0, stream>>>(
        obf, WoBf, (float*)d_out, SEQ, HID, TV, 0);
}
